// SNAttention_78168404787588
// MI455X (gfx1250) — compile-verified, matrix operand order checked by reading, not run on hardware
//
#include <hip/hip_runtime.h>

// ---------------------------------------------------------------------------
// SNAttention for MI455X / gfx1250 (wave32, WMMA bf16 16x16x32, TDM staging)
// B=4, C=256, CQK=32, N=64*64=4096
// ---------------------------------------------------------------------------

typedef __attribute__((ext_vector_type(16))) __bf16        v16bf;
typedef __attribute__((ext_vector_type(8)))  float         v8f;
typedef __attribute__((ext_vector_type(4)))  unsigned int  u32x4;
typedef __attribute__((ext_vector_type(8)))  int           i32x8;
typedef __attribute__((ext_vector_type(4)))  int           i32x4;

#define DEVINL __device__ __forceinline__

constexpr int B_   = 4;
constexpr int C_   = 256;
constexpr int CQK_ = 32;
constexpr int N_   = 4096;

constexpr int VSTRIDE = 40;   // LDS row stride (bf16 elems) for the V chunk: 32 data + 4-dword pad
constexpr int PSTRIDE = 36;   // LDS row stride (f32 elems) for the P tile:   32 data + pad

union FragBF { v16bf v; u32x4 q[2]; };

DEVINL v8f wmma_bf16(v16bf a, v16bf b, v8f c) {
  return __builtin_amdgcn_wmma_f32_16x16x32_bf16(false, a, false, b, (short)0, c,
                                                 false, false);
}

// Load a 16x32 bf16 A-style fragment (also used for B via B^T rows) from
// row-major memory with row stride `stride` (elements).
// Lane layout (ISA 7.12.2): lanes 0-15 row=l, K 0..7 & 16..23; lanes 16-31 K 8..15 & 24..31.
DEVINL v16bf load_frag(const __bf16* base, int row0, int stride, int lane) {
  const int l = lane & 15, hi = lane >> 4;
  const __bf16* p = base + (size_t)(row0 + l) * stride + hi * 8;
  FragBF f;
  f.q[0] = *(const u32x4*)(p);
  f.q[1] = *(const u32x4*)(p + 16);
  return f.v;
}

DEVINL float halfmax16(float t) {   // max over each 16-lane half (row of a C/D vgpr)
  t = fmaxf(t, __shfl_xor(t, 1, 32));
  t = fmaxf(t, __shfl_xor(t, 2, 32));
  t = fmaxf(t, __shfl_xor(t, 4, 32));
  t = fmaxf(t, __shfl_xor(t, 8, 32));
  return t;
}
DEVINL float halfsum16(float t) {
  t += __shfl_xor(t, 1, 32);
  t += __shfl_xor(t, 2, 32);
  t += __shfl_xor(t, 4, 32);
  t += __shfl_xor(t, 8, 32);
  return t;
}

// ---------------------------------------------------------------------------
// Kernel 1: convert weights to bf16
// ---------------------------------------------------------------------------
__global__ __launch_bounds__(256) void k_cvt_w(const float* Wq, const float* Wk,
                                               const float* Wv, __bf16* wq,
                                               __bf16* wk, __bf16* wv) {
  int i = blockIdx.x * 256 + threadIdx.x;
  if (i < CQK_ * C_) { wq[i] = (__bf16)Wq[i]; wk[i] = (__bf16)Wk[i]; }
  if (i < C_ * C_)   { wv[i] = (__bf16)Wv[i]; }
}

// ---------------------------------------------------------------------------
// Kernel 2: x [B][C][N] f32  ->  xT [B][N][C] bf16  (LDS tiled transpose)
// ---------------------------------------------------------------------------
__global__ __launch_bounds__(256) void k_xpose(const float* __restrict__ x,
                                               __bf16* __restrict__ xT) {
  __shared__ float tile[32][33];
  const int b  = blockIdx.z;
  const int c0 = blockIdx.y * 32;
  const int n0 = blockIdx.x * 32;
  const int tx = threadIdx.x & 31;
  const int ty = threadIdx.x >> 5;  // 0..7
  const float* xp = x + ((size_t)b * C_ + c0) * N_ + n0;
#pragma unroll
  for (int i = 0; i < 4; ++i) {
    int r = ty + i * 8;
    tile[r][tx] = xp[(size_t)r * N_ + tx];
  }
  __syncthreads();
  __bf16* xtp = xT + ((size_t)b * N_ + n0) * C_ + c0;
#pragma unroll
  for (int i = 0; i < 4; ++i) {
    int r = ty + i * 8;
    xtp[(size_t)r * C_ + tx] = (__bf16)tile[tx][r];
  }
}

// ---------------------------------------------------------------------------
// Kernel 3: q / kT projections. One wave = one 16-pixel tile, full 32 dims.
// q[b][n][d], kT[b][n][d]  (identical layouts; kT[m][d] == k[d][m])
// ---------------------------------------------------------------------------
__global__ __launch_bounds__(256) void k_proj_qk(const __bf16* __restrict__ xT,
                                                 const __bf16* __restrict__ wq,
                                                 const __bf16* __restrict__ wk,
                                                 const float* __restrict__ bq,
                                                 const float* __restrict__ bk,
                                                 __bf16* __restrict__ q,
                                                 __bf16* __restrict__ kT) {
  const int lane = threadIdx.x & 31;
  const int wid  = threadIdx.x >> 5;
  const int gw   = blockIdx.x * 8 + wid;   // 1024 waves total
  const int b    = gw >> 8;                // 256 n-tiles per batch
  const int n0   = (gw & 255) * 16;
  const __bf16* xtb = xT + (size_t)b * N_ * C_;

  v8f aq[2] = {}, ak[2] = {};
#pragma unroll 1
  for (int c0 = 0; c0 < C_; c0 += 32) {
    v16bf A = load_frag(xtb + c0, n0, C_, lane);
#pragma unroll
    for (int t = 0; t < 2; ++t) {
      v16bf Bq = load_frag(wq + c0, t * 16, C_, lane);  // B^T rows = Wq rows
      v16bf Bk = load_frag(wk + c0, t * 16, C_, lane);
      aq[t] = wmma_bf16(A, Bq, aq[t]);
      ak[t] = wmma_bf16(A, Bk, ak[t]);
    }
  }
  const int l = lane & 15, hi = lane >> 4;
#pragma unroll
  for (int t = 0; t < 2; ++t) {
    float bqv = bq[t * 16 + l];
    float bkv = bk[t * 16 + l];
#pragma unroll
    for (int r = 0; r < 8; ++r) {
      size_t idx = ((size_t)b * N_ + n0 + r + 8 * hi) * CQK_ + t * 16 + l;
      q[idx]  = (__bf16)(aq[t][r] + bqv);
      kT[idx] = (__bf16)(ak[t][r] + bkv);
    }
  }
}

// ---------------------------------------------------------------------------
// Kernel 4: v projection into [B][C][N] bf16 (rows = channels -> contiguous
// B^T rows for the PV gemm). One wave = one 16x16 (d x n) tile.
// ---------------------------------------------------------------------------
__global__ __launch_bounds__(256) void k_proj_v(const __bf16* __restrict__ xT,
                                                const __bf16* __restrict__ wv,
                                                const float* __restrict__ bv,
                                                __bf16* __restrict__ v) {
  const int lane = threadIdx.x & 31;
  const int wid  = threadIdx.x >> 5;
  const int gw   = blockIdx.x * 8 + wid;   // 16384 waves total
  const int b    = gw >> 12;               // 4096 tiles per batch
  const int rem  = gw & 4095;
  const int d0   = (rem >> 8) * 16;
  const int n0   = (rem & 255) * 16;
  const __bf16* xtb = xT + (size_t)b * N_ * C_;

  v8f acc = {};
#pragma unroll 1
  for (int c0 = 0; c0 < C_; c0 += 32) {
    v16bf A  = load_frag(wv + c0, d0, C_, lane);   // Wv rows (d, K=c)
    v16bf Bx = load_frag(xtb + c0, n0, C_, lane);  // B^T rows = xT rows
    acc = wmma_bf16(A, Bx, acc);
  }
  const int l = lane & 15, hi = lane >> 4;
#pragma unroll
  for (int r = 0; r < 8; ++r) {
    int d = d0 + r + 8 * hi;
    v[((size_t)b * C_ + d) * N_ + n0 + l] = (__bf16)(acc[r] + bv[d]);
  }
}

// ---------------------------------------------------------------------------
// Kernel 5: flash attention + residual.
// Block = 256 threads = 8 waves; wave (w) handles q-rows n0..n0+15 (w>>1)
// and channel half (w&1)*128..+127. V chunk (32 keys x 256 ch, 16 KB) staged
// into LDS via the Tensor Data Mover (double buffered, TENSORcnt-waited).
// ---------------------------------------------------------------------------
#if defined(__HIP_DEVICE_COMPILE__) && __has_builtin(__builtin_amdgcn_tensor_load_to_lds) && __has_builtin(__builtin_amdgcn_s_wait_tensorcnt)
#define USE_TDM 1
#endif

#ifdef USE_TDM
DEVINL void tdm_load_vchunk(const __bf16* gsrc, __bf16* lds_dst) {
  unsigned long long ga = (unsigned long long)(uintptr_t)gsrc;
  unsigned int lds = (unsigned int)(uintptr_t)lds_dst;  // low 32 bits = LDS offset
  // D# group0: count=1 | lds_addr | global_addr(57b) | type=2
  u32x4 g0;
  g0[0] = 1u;
  g0[1] = lds;
  g0[2] = (unsigned int)(ga & 0xFFFFFFFFu);
  g0[3] = (unsigned int)((ga >> 32) & 0x01FFFFFFu) | (2u << 30);
  // D# group1: data_size=2B, pad 4 dwords every 16 dwords (row 64B -> stride 80B)
  union { unsigned long long q[4]; i32x8 v; } g1;
  g1.q[0] = ((unsigned long long)1u << 16)     // data_size = 2B
          | (1ull << 20)                       // pad_enable
          | (3ull << 22)                       // pad_interval: 16 dwords
          | (3ull << 25)                       // pad_amount: 4 dwords
          | ((unsigned long long)(N_ & 0xFFFF) << 48);   // tensor_dim0 lo16
  g1.q[1] = (unsigned long long)(N_ >> 16)               // tensor_dim0 hi16
          | ((unsigned long long)C_ << 16)               // tensor_dim1
          | (32ull << 48);                               // tile_dim0 = 32 keys
  g1.q[2] = (unsigned long long)C_                       // tile_dim1 = 256 channels
          | ((unsigned long long)N_ << 32);              // tensor_dim0_stride lo32
  g1.q[3] = 0;                                           // stride hi bits, dim1_stride
  i32x4 z4 = {0, 0, 0, 0};
  i32x8 z8 = {0, 0, 0, 0, 0, 0, 0, 0};
  // clang-23 / therock-10.0 form: 6 args (g0, g1, g2, g3, extra, cpol)
  __builtin_amdgcn_tensor_load_to_lds(g0, g1.v, z4, z4, z8, 0);
}
#endif

__global__ __launch_bounds__(256, 1) void k_attn(const __bf16* __restrict__ q,
                                                 const __bf16* __restrict__ kT,
                                                 const __bf16* __restrict__ v,
                                                 const float* __restrict__ x,
                                                 float* __restrict__ out) {
  __shared__ __align__(16) __bf16 vbuf[2][C_ * VSTRIDE];  // 2 x 20 KB
  __shared__ __align__(16) float  pbuf[8][16 * PSTRIDE];  // 18 KB

  const int lane = threadIdx.x & 31;
  const int wid  = threadIdx.x >> 5;
  const int l    = lane & 15;
  const int hi   = lane >> 4;

  const int b     = blockIdx.x >> 6;                       // 64 blocks per batch
  const int n0    = (blockIdx.x & 63) * 64 + (wid >> 1) * 16;
  const int dbase = (wid & 1) * 128;

  const __bf16* qb = q  + (size_t)b * N_ * CQK_;
  const __bf16* kb = kT + (size_t)b * N_ * CQK_;
  const __bf16* vb = v  + (size_t)b * C_ * N_;

  const v16bf Aq = load_frag(qb, n0, CQK_, lane);          // 16 rows x K=32 (full CQK)

  v8f acc[8] = {};
  float mrow[8], lrow[8];
#pragma unroll
  for (int r = 0; r < 8; ++r) { mrow[r] = -1e30f; lrow[r] = 0.0f; }

  // ---- stage chunk 0 ----
#ifdef USE_TDM
  if (wid == 0) {
    tdm_load_vchunk(vb, vbuf[0]);
    __builtin_amdgcn_s_wait_tensorcnt(0);
  }
#else
  {
    int row = threadIdx.x;  // 256 threads = 256 channel rows
    const __bf16* src = vb + (size_t)row * N_;
    __bf16* dst = &vbuf[0][row * VSTRIDE];
#pragma unroll
    for (int j = 0; j < 4; ++j)
      *(u32x4*)(dst + j * 8) = *(const u32x4*)(src + j * 8);
  }
#endif
  __syncthreads();

  constexpr int NITER = N_ / 32;  // 128
#pragma unroll 1
  for (int it = 0; it < NITER; ++it) {
    const int m0 = it * 32;
    // ---- prefetch chunk it+1 into the other buffer ----
    if (it + 1 < NITER) {
#ifdef USE_TDM
      if (wid == 0) tdm_load_vchunk(vb + (m0 + 32), vbuf[(it + 1) & 1]);
#else
      int row = threadIdx.x;
      const __bf16* src = vb + (size_t)row * N_ + (m0 + 32);
      __bf16* dst = &vbuf[(it + 1) & 1][row * VSTRIDE];
#pragma unroll
      for (int j = 0; j < 4; ++j)
        *(u32x4*)(dst + j * 8) = *(const u32x4*)(src + j * 8);
#endif
    }
    const __bf16* vchunk = vbuf[it & 1];

    // ---- energy: S = q . k^T for 32 key columns (2 WMMAs, K=32 full) ----
    v16bf Bk0 = load_frag(kb, m0,      CQK_, lane);
    v16bf Bk1 = load_frag(kb, m0 + 16, CQK_, lane);
    v8f zero = {};
    v8f S0 = wmma_bf16(Aq, Bk0, zero);
    v8f S1 = wmma_bf16(Aq, Bk1, zero);

    // ---- online softmax (row r+8*hi lives in vgpr r, one 16-lane half) ----
    float sc[8];
    float* pb = pbuf[wid];
#pragma unroll
    for (int r = 0; r < 8; ++r) {
      float t = halfmax16(fmaxf(S0[r], S1[r]));
      float mnew = fmaxf(mrow[r], t);
      float p0 = __expf(S0[r] - mnew);
      float p1 = __expf(S1[r] - mnew);
      float rs = halfsum16(p0 + p1);
      float s  = __expf(mrow[r] - mnew);
      lrow[r] = lrow[r] * s + rs;
      mrow[r] = mnew;
      sc[r]   = s;
      pb[(r + 8 * hi) * PSTRIDE + l]      = p0;
      pb[(r + 8 * hi) * PSTRIDE + l + 16] = p1;
    }
#pragma unroll
    for (int t = 0; t < 8; ++t) {
      v8f a = acc[t];
#pragma unroll
      for (int r = 0; r < 8; ++r) a[r] *= sc[r];
      acc[t] = a;
    }

    // ---- P: C/D layout -> A fragment via LDS (wave-private, DS in-order) ----
    const int cb0 = hi * 8;
    const float* prow = pb + l * PSTRIDE + cb0;
    float fv[16];
    *(float4*)&fv[0]  = *(const float4*)(prow);
    *(float4*)&fv[4]  = *(const float4*)(prow + 4);
    *(float4*)&fv[8]  = *(const float4*)(prow + 16);
    *(float4*)&fv[12] = *(const float4*)(prow + 20);
    v16bf ap;
#pragma unroll
    for (int i = 0; i < 16; ++i) ap[i] = (__bf16)fv[i];

    // ---- PV: acc[t] += P[16x32] x V^T[32 x 16] for 8 channel tiles ----
#pragma unroll
    for (int t = 0; t < 8; ++t) {
      const int d0 = dbase + t * 16;
      const __bf16* vrow = vchunk + (d0 + l) * VSTRIDE + hi * 8;
      FragBF Bv;
      Bv.q[0] = *(const u32x4*)(vrow);
      Bv.q[1] = *(const u32x4*)(vrow + 16);
      acc[t] = wmma_bf16(ap, Bv.v, acc[t]);
    }

#ifdef USE_TDM
    if (wid == 0 && it + 1 < NITER) __builtin_amdgcn_s_wait_tensorcnt(0);
#endif
    __syncthreads();
  }

  // ---- epilogue: 1/l scaling + residual ----
  float inv[8];
#pragma unroll
  for (int r = 0; r < 8; ++r) inv[r] = 1.0f / lrow[r];
  const float* xb = x + (size_t)b * C_ * N_;
  float* ob = out + (size_t)b * C_ * N_;
#pragma unroll
  for (int t = 0; t < 8; ++t) {
    int d = dbase + t * 16 + l;
#pragma unroll
    for (int r = 0; r < 8; ++r) {
      size_t idx = (size_t)d * N_ + n0 + r + 8 * hi;
      ob[idx] = acc[t][r] * inv[r] + xb[idx];
    }
  }
}

// ---------------------------------------------------------------------------
// Host launcher
// ---------------------------------------------------------------------------
extern "C" void kernel_launch(void* const* d_in, const int* in_sizes, int n_in,
                              void* d_out, int out_size, void* d_ws, size_t ws_size,
                              hipStream_t stream) {
  (void)in_sizes; (void)n_in; (void)out_size; (void)ws_size;
  const float* x  = (const float*)d_in[0];
  const float* Wq = (const float*)d_in[1];
  const float* bq = (const float*)d_in[2];
  const float* Wk = (const float*)d_in[3];
  const float* bk = (const float*)d_in[4];
  const float* Wv = (const float*)d_in[5];
  const float* bv = (const float*)d_in[6];
  float* out = (float*)d_out;

  char* ws = (char*)d_ws;
  size_t off = 0;
  auto carve = [&](size_t bytes) -> char* {
    char* p = ws + off;
    off = (off + bytes + 255) & ~(size_t)255;
    return p;
  };
  __bf16* xT  = (__bf16*)carve((size_t)B_ * N_ * C_ * 2);    // 8 MB
  __bf16* qb  = (__bf16*)carve((size_t)B_ * N_ * CQK_ * 2);  // 1 MB
  __bf16* kb  = (__bf16*)carve((size_t)B_ * N_ * CQK_ * 2);  // 1 MB
  __bf16* vbf = (__bf16*)carve((size_t)B_ * C_ * N_ * 2);    // 8 MB
  __bf16* wqb = (__bf16*)carve((size_t)CQK_ * C_ * 2);
  __bf16* wkb = (__bf16*)carve((size_t)CQK_ * C_ * 2);
  __bf16* wvb = (__bf16*)carve((size_t)C_ * C_ * 2);

  k_cvt_w<<<(C_ * C_) / 256, 256, 0, stream>>>(Wq, Wk, Wv, wqb, wkb, wvb);
  k_xpose<<<dim3(N_ / 32, C_ / 32, B_), 256, 0, stream>>>(x, xT);
  k_proj_qk<<<(B_ * (N_ / 16)) / 8, 256, 0, stream>>>(xT, wqb, wkb, bq, bk, qb, kb);
  k_proj_v<<<(B_ * (C_ / 16) * (N_ / 16)) / 8, 256, 0, stream>>>(xT, wvb, bv, vbf);
  k_attn<<<B_ * (N_ / 64), 256, 0, stream>>>(qb, kb, vbf, x, out);
}
